// ProteinFeatures_16690242913059
// MI455X (gfx1250) — matrix operand order repaired and checked by reading, not compiled, not run on hardware
//
#include <hip/hip_runtime.h>
#include <math.h>
#include <float.h>

// Problem constants (from reference)
#define NB 4
#define NL 1024
#define NK 30
#define NCH 128
#define KPE 448          // 434 edge features padded to multiple of 32
#define KPN 32           // 22 node features padded to 32
#define NEDGE (NB*NL*NK) // 122880
#define EDGE_COLS 434
#define OUT_VN 0
#define OUT_E  (NB*NL*NCH)              // 524288
#define OUT_EIDX (OUT_E + NEDGE*NCH)    // 16252928

typedef _Float16 v16h __attribute__((ext_vector_type(16)));
typedef _Float16 v8h  __attribute__((ext_vector_type(8)));
typedef float    v8f  __attribute__((ext_vector_type(8)));

// PAIRS atom indices: N=0 Ca=1 C=2 O=3 Cb=4
__constant__ int PA_tab[24] = {0,2,3,4,1,1,1,1,0,0,0,4,4,3,0,2,3,4,2,3,4,2,3,2};
__constant__ int PB_tab[24] = {0,2,3,4,0,2,3,4,2,3,4,2,3,2,1,1,1,1,0,0,0,4,4,3};

struct F3 { float x, y, z; };
__device__ inline F3 operator-(F3 a, F3 b){ return {a.x-b.x, a.y-b.y, a.z-b.z}; }
__device__ inline F3 operator+(F3 a, F3 b){ return {a.x+b.x, a.y+b.y, a.z+b.z}; }
__device__ inline F3 operator*(float s, F3 a){ return {s*a.x, s*a.y, s*a.z}; }
__device__ inline float dot3(F3 a, F3 b){ return a.x*b.x + a.y*b.y + a.z*b.z; }
__device__ inline F3 cross3(F3 a, F3 b){
  return { a.y*b.z - a.z*b.y, a.z*b.x - a.x*b.z, a.x*b.y - a.y*b.x };
}
__device__ inline F3 norm3(F3 a){
  float n = sqrtf(dot3(a,a));
  if (n > 0.f){ float inv = 1.f/n; return {a.x*inv, a.y*inv, a.z*inv}; }
  return {0.f, 0.f, 0.f};
}
__device__ inline float sgn(float x){ return (x > 0.f) ? 1.f : ((x < 0.f) ? -1.f : 0.f); }

__device__ inline F3 ldAtom(const float* X, int b, int i, int a){
  const float* p = X + ((size_t)((b*NL + i)*4 + a))*3;
  return {p[0], p[1], p[2]};
}
__device__ inline F3 ldBB(const float* X, int b, int m){   // backbone chain index m in [0,3L)
  return ldAtom(X, b, m/3, m%3);
}
__device__ inline F3 atom5(const float* X, const float* CbB, int b, int i, int a){
  if (a < 4) return ldAtom(X, b, i, a);
  const float* p = CbB + (size_t)(b*NL + i)*3;
  return {p[0], p[1], p[2]};
}

// ---------------------------------------------------------------------------
// Kernel 0: convert + transpose weights to N-major f16 (B-fragment friendly)
// BwTe[n*KPE + k] = W_edge[k*128 + n] (zero padded), similarly node.
// ---------------------------------------------------------------------------
__global__ void conv_w_kernel(const float* We, const float* Wn,
                              _Float16* BwTe, _Float16* BwTn){
  int idx = blockIdx.x*256 + threadIdx.x;
  if (idx < NCH*KPE){
    int n = idx / KPE, kk = idx % KPE;
    BwTe[idx] = (kk < EDGE_COLS) ? (_Float16)We[kk*NCH + n] : (_Float16)0.f;
  }
  if (idx < NCH*KPN){
    int n = idx / KPN, kk = idx % KPN;
    BwTn[idx] = (kk < 22) ? (_Float16)Wn[kk*NCH + n] : (_Float16)0.f;
  }
}

// ---------------------------------------------------------------------------
// Kernel 1: per-node prep: Cb virtual atom + local frame Qm (3x3, row-major)
// ---------------------------------------------------------------------------
__global__ void prep_kernel(const float* X, float* CbB, float* QmB){
  int idx = blockIdx.x*blockDim.x + threadIdx.x;
  if (idx >= NB*NL) return;
  int b = idx / NL, n = idx % NL;
  F3 Na = ldAtom(X,b,n,0), Ca = ldAtom(X,b,n,1), Cc = ldAtom(X,b,n,2);
  F3 bv = Ca - Na, cv = Cc - Ca;
  F3 av = cross3(bv, cv);
  F3 Cb = (-0.58273431f)*av + 0.56802827f*bv + (-0.54067466f)*cv + Ca;
  float* cb = CbB + (size_t)idx*3;
  cb[0]=Cb.x; cb[1]=Cb.y; cb[2]=Cb.z;
  float* Q = QmB + (size_t)idx*9;
  if (n < NL-1){
    F3 u0 = norm3(Ca - Na);
    F3 u1 = norm3(Cc - Ca);
    F3 b1 = norm3(u0 - u1);
    F3 n0 = norm3(cross3(u0, u1));
    F3 r2 = cross3(b1, n0);
    Q[0]=b1.x; Q[1]=b1.y; Q[2]=b1.z;
    Q[3]=n0.x; Q[4]=n0.y; Q[5]=n0.z;
    Q[6]=r2.x; Q[7]=r2.y; Q[8]=r2.z;
  } else {
    #pragma unroll
    for (int t=0;t<9;t++) Q[t] = 0.f;
  }
}

// ---------------------------------------------------------------------------
// Kernel 2: masked top-K neighbor search. One 256-thread WG per (b,i) row.
// Produces E_idx (int32, into d_out tail) and D_neighbors (workspace).
// mask==0 rows fall back to |residue offset| keys (matches the jnp.where).
// ---------------------------------------------------------------------------
__global__ void topk_kernel(const float* X, const float* mask, const int* res,
                            float* Dnb, int* Eidx){
  __shared__ float dist[NL];
  __shared__ float rv[256];
  __shared__ int   ridx[256];
  int bi = blockIdx.x;            // b*NL + i
  int b = bi / NL, i = bi % NL;
  int tid = threadIdx.x;
  F3 Cai = ldAtom(X,b,i,1);
  float mi = mask[bi];
  int ri_res = res[bi];

  float lmax = -FLT_MAX;
  #pragma unroll
  for (int q=0;q<4;q++){
    int j = tid + q*256;
    F3 Caj = ldAtom(X,b,j,1);
    float mj = mask[b*NL + j];
    F3 d = Cai - Caj;
    float D = mi*mj*sqrtf(dot3(d,d) + 1e-6f);
    dist[j] = D;
    lmax = fmaxf(lmax, D);
  }
  rv[tid] = lmax; __syncthreads();
  for (int off=128; off; off>>=1){
    if (tid < off) rv[tid] = fmaxf(rv[tid], rv[tid+off]);
    __syncthreads();
  }
  float Dmax = rv[0];
  __syncthreads();
  #pragma unroll
  for (int q=0;q<4;q++){
    int j = tid + q*256;
    float key;
    if (mi == 0.f){
      int dsq = ri_res - res[b*NL + j];
      key = (float)(dsq < 0 ? -dsq : dsq);
    } else {
      float mj = mask[b*NL + j];
      key = dist[j] + (1.f - mi*mj)*Dmax;
    }
    dist[j] = key;
  }
  __syncthreads();

  for (int kk=0; kk<NK; kk++){
    float bvv = FLT_MAX; int bj = 0;
    #pragma unroll
    for (int q=0;q<4;q++){
      int j = tid + q*256;
      float v = dist[j];
      if (v < bvv){ bvv = v; bj = j; }
    }
    rv[tid] = bvv; ridx[tid] = bj;
    __syncthreads();
    for (int off=128; off; off>>=1){
      if (tid < off){
        if (rv[tid+off] < rv[tid] ||
            (rv[tid+off] == rv[tid] && ridx[tid+off] < ridx[tid])){
          rv[tid] = rv[tid+off]; ridx[tid] = ridx[tid+off];
        }
      }
      __syncthreads();
    }
    if (tid == 0){
      int e = bi*NK + kk;
      Eidx[e] = ridx[0];
      Dnb[e]  = rv[0];
      dist[ridx[0]] = FLT_MAX;
    }
    __syncthreads();
  }
}

// ---------------------------------------------------------------------------
// Shared WMMA GEMM(16 x KP x 128) + LayerNorm epilogue for a 16-row tile in
// LDS. 8 waves, each wave owns a 16-wide N slice. f16 A/B, f32 accumulate.
// ---------------------------------------------------------------------------
__device__ inline void wmma_gemm_ln(const _Float16* ftile, int KP,
                                    const _Float16* BwT, const float* gam,
                                    const float* bet, float* otile,
                                    float* outBase, int tileRow0, int tid){
  int wave = tid >> 5, lane = tid & 31;
  int nbase = wave*16;
  int mloc  = lane & 15;
  int hsel  = lane >> 4;
  v8f acc = {};
  const _Float16* arow = ftile + mloc*KP;
  const _Float16* brow0 = BwT + (size_t)(nbase + mloc)*KP;
  for (int k0 = 0; k0 < KP; k0 += 32){
    int kb = hsel*8;
    v8h alo = *(const v8h*)(arow + k0 + kb);
    v8h ahi = *(const v8h*)(arow + k0 + 16 + kb);
    v16h afr;
    #pragma unroll
    for (int t=0;t<8;t++){ afr[t] = alo[t]; afr[t+8] = ahi[t]; }
    const _Float16* brow = brow0 + k0 + hsel*8;
    __builtin_prefetch(brow + 32, 0, 0);
    v8h blo = *(const v8h*)brow;
    v8h bhi = *(const v8h*)(brow + 16);
    v16h bfr;
    #pragma unroll
    for (int t=0;t<8;t++){ bfr[t] = blo[t]; bfr[t+8] = bhi[t]; }
    acc = __builtin_amdgcn_wmma_f32_16x16x32_f16(
        false, afr, false, bfr, (short)0, acc, false, false);
  }
  #pragma unroll
  for (int r=0;r<8;r++){
    int m = r + hsel*8;
    otile[m*NCH + nbase + mloc] = acc[r];
  }
  __syncthreads();
  // LayerNorm over N=128 per row: 16 threads/row, 8 cols/thread.
  int r2 = tid >> 4;
  int slot = tid & 15;
  const float* orow = otile + r2*NCH;
  float s1 = 0.f, s2 = 0.f;
  #pragma unroll
  for (int q=0;q<8;q++){ float x = orow[slot*8 + q]; s1 += x; s2 += x*x; }
  for (int off=8; off; off>>=1){
    s1 += __shfl_xor(s1, off, 16);
    s2 += __shfl_xor(s2, off, 16);
  }
  float mu  = s1*(1.f/128.f);
  float var = s2*(1.f/128.f) - mu*mu;
  float rstd = rsqrtf(var + 1e-5f);
  float* og = outBase + (size_t)(tileRow0 + r2)*NCH;
  #pragma unroll
  for (int q=0;q<8;q++){
    int n = slot*8 + q;
    og[n] = (orow[n] - mu)*rstd*gam[n] + bet[n];
  }
}

// ---------------------------------------------------------------------------
// Kernel 3: fused edge features (16 edges/WG, built in LDS as f16) + WMMA
// GEMM with W_edge + layernorm -> E output.
// Column map: [0,16) pos | [16,416) 25x16 RBF | [416,428) E_direct |
//             [428,432) quat | 432 chain | 433 mask_j | [434,448) zero pad
// ---------------------------------------------------------------------------
__global__ void __launch_bounds__(256)
edge_kernel(const float* X, const float* CbB, const float* QmB,
            const float* Dnb, const int* Eidx, const float* mask,
            const int* res, const int* chain, const float* Wpos,
            const float* bpos, const _Float16* BwT, const float* gam,
            const float* bet, float* outE){
  __shared__ __align__(16) _Float16 ftile[16*KPE];
  __shared__ float otile[16*NCH];
  int tid = threadIdx.x;
  int row = tid >> 4, s = tid & 15;
  int e = blockIdx.x*16 + row;
  int k = e % NK; (void)k;
  int bi = e / NK;
  int i = bi % NL, b = bi / NL;
  int j = Eidx[e];
  _Float16* frow = ftile + row*KPE;

  // RBF groups: group 0 = Ca topk distance, 1..24 = atom pairs at neighbor j
  for (int g5 = s; g5 < 25; g5 += 16){
    float D;
    if (g5 == 0) D = Dnb[e];
    else {
      F3 A  = atom5(X, CbB, b, i, PA_tab[g5-1]);
      F3 Bv = atom5(X, CbB, b, j, PB_tab[g5-1]);
      F3 d = A - Bv;
      D = sqrtf(dot3(d,d) + 1e-6f);
    }
    #pragma unroll
    for (int m=0;m<16;m++){
      float mu = 2.0f + (20.0f/15.0f)*(float)m;
      float t = (D - mu)*(1.0f/1.25f);
      frow[16 + g5*16 + m] = (_Float16)__expf(-t*t);
    }
  }
  if (s == 1){
    int Ech = (chain[b*NL+i] == chain[b*NL+j]) ? 1 : 0;
    int off = res[b*NL+i] - res[b*NL+j];
    int dc = off + 32; dc = dc < 0 ? 0 : (dc > 64 ? 64 : dc);
    int dd = Ech ? dc : 65;
    #pragma unroll
    for (int c=0;c<16;c++) frow[c] = (_Float16)(Wpos[dd*16 + c] + bpos[c]);
  }
  if (s == 2){
    #pragma unroll
    for (int c = EDGE_COLS; c < KPE; c++) frow[c] = (_Float16)0.f;
  }
  if (s == 0){
    const float* Qi = QmB + (size_t)(b*NL+i)*9;
    const float* Qj = QmB + (size_t)(b*NL+j)*9;
    F3 Ni = ldAtom(X,b,i,0);   // Xa = backbone atom 0 (N) per reference
    F3 nb[4] = { ldAtom(X,b,j,1), ldAtom(X,b,j,0), ldAtom(X,b,j,2), ldAtom(X,b,j,3) };
    #pragma unroll
    for (int a=0;a<4;a++){
      F3 v = nb[a] - Ni;
      F3 dU = { Qi[0]*v.x + Qi[1]*v.y + Qi[2]*v.z,
                Qi[3]*v.x + Qi[4]*v.y + Qi[5]*v.z,
                Qi[6]*v.x + Qi[7]*v.y + Qi[8]*v.z };
      F3 nd = norm3(dU);
      frow[416 + a*3 + 0] = (_Float16)nd.x;
      frow[416 + a*3 + 1] = (_Float16)nd.y;
      frow[416 + a*3 + 2] = (_Float16)nd.z;
    }
    // R = Qi^T * Qj ; quaternion features
    float R[3][3];
    #pragma unroll
    for (int r=0;r<3;r++)
      #pragma unroll
      for (int c=0;c<3;c++)
        R[r][c] = Qi[0+r]*Qj[0+c] + Qi[3+r]*Qj[3+c] + Qi[6+r]*Qj[6+c];
    float Rxx=R[0][0], Ryy=R[1][1], Rzz=R[2][2];
    float m0 = 0.5f*sqrtf(fabsf(1.f + Rxx - Ryy - Rzz));
    float m1 = 0.5f*sqrtf(fabsf(1.f - Rxx + Ryy - Rzz));
    float m2 = 0.5f*sqrtf(fabsf(1.f - Rxx - Ryy + Rzz));
    float q0 = sgn(R[2][1]-R[1][2])*m0;
    float q1 = sgn(R[0][2]-R[2][0])*m1;
    float q2 = sgn(R[1][0]-R[0][1])*m2;
    float w  = 0.5f*sqrtf(fmaxf(0.f, 1.f + Rxx + Ryy + Rzz));
    float qn = sqrtf(q0*q0 + q1*q1 + q2*q2 + w*w);
    float inv = (qn > 0.f) ? 1.f/qn : 0.f;
    frow[428] = (_Float16)(q0*inv);
    frow[429] = (_Float16)(q1*inv);
    frow[430] = (_Float16)(q2*inv);
    frow[431] = (_Float16)(w*inv);
    frow[432] = (_Float16)((chain[b*NL+i]==chain[b*NL+j]) ? 1.f : 0.f);
    frow[433] = (_Float16)mask[b*NL + j];
  }
  __syncthreads();
  wmma_gemm_ln(ftile, KPE, BwT, gam, bet, otile, outE, blockIdx.x*16, tid);
}

// ---------------------------------------------------------------------------
// Kernel 4: fused node features (dihedrals/angles/V_direct) + WMMA + LN -> Vn
// ---------------------------------------------------------------------------
__global__ void __launch_bounds__(256)
node_kernel(const float* X, const float* QmB, const float* mask,
            const _Float16* BwT, const float* gam, const float* bet,
            float* outV){
  __shared__ __align__(16) _Float16 ftile[16*KPN];
  __shared__ float otile[16*NCH];
  int tid = threadIdx.x;
  if (tid < 16){
    int idx = blockIdx.x*16 + tid;
    int b = idx / NL, n = idx % NL;
    _Float16* frow = ftile + tid*KPN;
    float mi = mask[idx];
    float cd[3], sd[3], caF[3], saF[3];
    #pragma unroll
    for (int slot=0; slot<3; slot++){
      int t = 3*n + slot - 1;
      float D = 0.f, A = 0.f;
      if (t >= 0 && t <= 3*NL - 4){
        F3 p0 = ldBB(X,b,t),   p1 = ldBB(X,b,t+1);
        F3 p2 = ldBB(X,b,t+2), p3 = ldBB(X,b,t+3);
        F3 u0 = norm3(p1-p0), u1 = norm3(p2-p1), u2 = norm3(p3-p2);
        F3 n0 = norm3(cross3(u0,u1));
        F3 n1 = norm3(cross3(u1,u2));
        float cosD = fminf(fmaxf(dot3(n0,n1), -1.f + 1e-7f), 1.f - 1e-7f);
        F3 v = norm3(cross3(n0,n1));
        D = sgn(-dot3(v,u1)) * acosf(cosD);
        float cosA = fminf(fmaxf(dot3(u0,u1), -1.f + 1e-7f), 1.f - 1e-7f);
        A = acosf(cosA);
      }
      cd[slot]=cosf(D); sd[slot]=sinf(D);
      caF[slot]=cosf(A); saF[slot]=sinf(A);
    }
    const float* Q = QmB + (size_t)idx*9;
    F3 Nn = ldAtom(X,b,n,0);
    F3 at[3] = { Nn, ldAtom(X,b,n,2), ldAtom(X,b,n,3) };
    float vd[9];
    #pragma unroll
    for (int a=0;a<3;a++){
      F3 v = at[a] - Nn;
      F3 dU = { Q[0]*v.x + Q[1]*v.y + Q[2]*v.z,
                Q[3]*v.x + Q[4]*v.y + Q[5]*v.z,
                Q[6]*v.x + Q[7]*v.y + Q[8]*v.z };
      F3 nd = norm3(dU);
      vd[a*3]=nd.x; vd[a*3+1]=nd.y; vd[a*3+2]=nd.z;
    }
    float f[22];
    f[0]=cd[0]; f[1]=cd[1]; f[2]=cd[2]; f[3]=sd[0]; f[4]=sd[1]; f[5]=sd[2];
    f[6]=caF[0]; f[7]=caF[1]; f[8]=caF[2]; f[9]=saF[0]; f[10]=saF[1]; f[11]=saF[2];
    #pragma unroll
    for (int c=0;c<12;c++) f[c] *= mi;          // V_angles * mask
    #pragma unroll
    for (int c=0;c<9;c++)  f[12+c] = vd[c]*mi;  // V_direct * mask
    f[21] = mi;
    #pragma unroll
    for (int c=0;c<22;c++) frow[c] = (_Float16)f[c];
    #pragma unroll
    for (int c=22;c<KPN;c++) frow[c] = (_Float16)0.f;
  }
  __syncthreads();
  wmma_gemm_ln(ftile, KPN, BwT, gam, bet, otile, outV, blockIdx.x*16, tid);
}

// ---------------------------------------------------------------------------
extern "C" void kernel_launch(void* const* d_in, const int* in_sizes, int n_in,
                              void* d_out, int out_size, void* d_ws, size_t ws_size,
                              hipStream_t stream){
  (void)in_sizes; (void)n_in; (void)out_size; (void)ws_size;
  const float* X     = (const float*)d_in[0];
  const float* mask  = (const float*)d_in[1];
  const int*   res   = (const int*)  d_in[2];
  const int*   chain = (const int*)  d_in[3];
  const float* Wpos  = (const float*)d_in[4];
  const float* bpos  = (const float*)d_in[5];
  const float* Wedge = (const float*)d_in[6];
  const float* Wnode = (const float*)d_in[7];
  const float* ge    = (const float*)d_in[8];
  const float* be    = (const float*)d_in[9];
  const float* gn    = (const float*)d_in[10];
  const float* bn    = (const float*)d_in[11];

  float* out  = (float*)d_out;
  float* outV = out + OUT_VN;
  float* outE = out + OUT_E;
  int*   outI = ((int*)d_out) + OUT_EIDX;  // E_idx (int32 bits) in output tail

  // Workspace layout (~0.8 MB total)
  float*    CbB  = (float*)d_ws;              // 4*1024*3
  float*    QmB  = CbB + NB*NL*3;             // 4*1024*9
  float*    Dnb  = QmB + NB*NL*9;             // 4*1024*30
  _Float16* BwTe = (_Float16*)(Dnb + NEDGE);  // 128*448
  _Float16* BwTn = BwTe + NCH*KPE;            // 128*32

  conv_w_kernel<<<(NCH*KPE + 255)/256, 256, 0, stream>>>(Wedge, Wnode, BwTe, BwTn);
  prep_kernel<<<(NB*NL + 255)/256, 256, 0, stream>>>(X, CbB, QmB);
  topk_kernel<<<NB*NL, 256, 0, stream>>>(X, mask, res, Dnb, outI);
  edge_kernel<<<NEDGE/16, 256, 0, stream>>>(X, CbB, QmB, Dnb, outI, mask, res,
                                            chain, Wpos, bpos, BwTe, ge, be, outE);
  node_kernel<<<(NB*NL)/16, 256, 0, stream>>>(X, QmB, mask, BwTn, gn, bn, outV);
}
